// MambaSequenceClassificationSP_23261542875727
// MI455X (gfx1250) — matrix operand order repaired and checked by reading, not compiled, not run on hardware
//
#include <hip/hip_runtime.h>
#include <hip/hip_bf16.h>
#include <math.h>

#define BATCH    8
#define SEQLEN   1024
#define DMODEL   256
#define NLAYERS  2
#define DSTATE   64
#define DCONV    4
#define HEADDIM  64
#define DINNER   512
#define NHEADS   8
#define CONVDIM  640        // DINNER + 2*DSTATE
#define DINPROJ  1160       // 2*DINNER + 2*DSTATE + NHEADS
#define ROWS     (BATCH*SEQLEN)
#define EPSV     1e-5f

typedef __attribute__((ext_vector_type(16))) __bf16 v16bf;
typedef __attribute__((ext_vector_type(8)))  float  v8f;

union BFrag { v16bf v; uint4 q[2]; };
union FAcc  { v8f  v; float f[8]; };

__device__ __forceinline__ unsigned short f2bf_bits(float f) {
  unsigned u = __float_as_uint(f);
  u += 0x7FFFu + ((u >> 16) & 1u);         // round-to-nearest-even
  return (unsigned short)(u >> 16);
}
__device__ __forceinline__ float siluf(float x) { return x / (1.f + expf(-x)); }
__device__ __forceinline__ float geluf(float x) { return 0.5f * x * (1.f + erff(x * 0.70710678118654752f)); }

// ---------------- embedding (+ bf16 shadow of x) ----------------
__global__ void k_embed(const int* __restrict__ ids, const float* __restrict__ emb,
                        float* __restrict__ x, unsigned short* __restrict__ xb) {
  int idx = blockIdx.x * blockDim.x + threadIdx.x;          // ROWS*DMODEL
  int row = idx >> 8, d = idx & 255;
  float v = emb[(size_t)ids[row] * DMODEL + d];
  x[idx] = v;
  xb[idx] = f2bf_bits(v);
}

// ---------------- generic f32 -> bf16 ----------------
__global__ void k_f32_to_bf16(const float* __restrict__ src, unsigned short* __restrict__ dst, int n) {
  int idx = blockIdx.x * blockDim.x + threadIdx.x;
  if (idx < n) dst[idx] = f2bf_bits(src[idx]);
}

// ---------------- bf16 WMMA GEMM:  C[M,N] (+= addsrc) = A[M,K] * Bw[N,K]^T ----------------
// block tile 128x64, 256 threads = 8 waves (4 along M x 2 along N),
// each wave owns a 32x32 output tile = 2x2 WMMA accumulators; K step 32.
__global__ __launch_bounds__(256)
void k_gemm_bf16(const unsigned short* __restrict__ A, const unsigned short* __restrict__ Bw,
                 const float* __restrict__ addsrc, float* __restrict__ C,
                 int M, int N, int K, int ldc) {
  __shared__ __align__(16) unsigned short As[128][32];   // 8 KB
  __shared__ __align__(16) unsigned short Bs[64][32];    // 4 KB

  const int tid  = threadIdx.x;
  const int lane = tid & 31;
  const int wave = tid >> 5;
  const int wm   = wave & 3;          // 0..3  (32-row strip)
  const int wn   = wave >> 2;         // 0..1  (32-col strip)
  const int m0   = blockIdx.y * 128;
  const int n0   = blockIdx.x * 64;

  v8f acc00 = {}; v8f acc01 = {}; v8f acc10 = {}; v8f acc11 = {};

  // staging: A 128x32 (two uint4 per thread), B 64x32 (one uint4 per thread)
  const int ra = tid >> 1, sa = (tid & 1) * 16;
  const int rb = tid >> 2, sb = (tid & 3) * 8;

  const int kTiles = K >> 5;
  for (int kt = 0; kt < kTiles; ++kt) {
    const size_t kaBase = (size_t)(m0 + ra) * K + kt * 32 + sa;
    *(uint4*)&As[ra][sa]     = *(const uint4*)&A[kaBase];
    *(uint4*)&As[ra][sa + 8] = *(const uint4*)&A[kaBase + 8];
    const int bn = n0 + rb;
    uint4 bv = {0u, 0u, 0u, 0u};
    if (bn < N) bv = *(const uint4*)&Bw[(size_t)bn * K + kt * 32 + sb];
    *(uint4*)&Bs[rb][sb] = bv;
    if (kt + 1 < kTiles)
      __builtin_prefetch(&A[(size_t)(m0 + ra) * K + (kt + 1) * 32 + sa], 0, 1);
    __syncthreads();

    // ---- fragments per ISA 16-bit lane packing ----
    // A 16x32: lanes 0-15 -> K[0..7],K[16..23]; lanes 16-31 -> K[8..15],K[24..31]
    const int fr = lane & 15;
    const int hi = (lane >> 4) * 8;
    BFrag a0, a1;
    a0.q[0] = *(const uint4*)&As[wm * 32 + fr][hi];
    a0.q[1] = *(const uint4*)&As[wm * 32 + fr][16 + hi];
    a1.q[0] = *(const uint4*)&As[wm * 32 + 16 + fr][hi];
    a1.q[1] = *(const uint4*)&As[wm * 32 + 16 + fr][16 + hi];
    // B 32x16: lanes 0-15 col=lane K[0..7],K[16..23]; lanes 16-31 col=lane-16 K[8..15],K[24..31]
    const int kb = (lane >> 4) * 8;
    BFrag b0, b1;
    b0.q[0] = *(const uint4*)&Bs[wn * 32 + fr][kb];
    b0.q[1] = *(const uint4*)&Bs[wn * 32 + fr][kb + 16];
    b1.q[0] = *(const uint4*)&Bs[wn * 32 + 16 + fr][kb];
    b1.q[1] = *(const uint4*)&Bs[wn * 32 + 16 + fr][kb + 16];

    acc00 = __builtin_amdgcn_wmma_f32_16x16x32_bf16(false, a0.v, false, b0.v, (short)0, acc00, false, false);
    acc01 = __builtin_amdgcn_wmma_f32_16x16x32_bf16(false, a0.v, false, b1.v, (short)0, acc01, false, false);
    acc10 = __builtin_amdgcn_wmma_f32_16x16x32_bf16(false, a1.v, false, b0.v, (short)0, acc10, false, false);
    acc11 = __builtin_amdgcn_wmma_f32_16x16x32_bf16(false, a1.v, false, b1.v, (short)0, acc11, false, false);
    __syncthreads();
  }

  // ---- store: VGPR r -> M = r (+8 for lanes 16-31), col = lane&15 ----
  const int rowb0 = m0 + wm * 32 + ((lane >> 4) * 8);      // first 16-row tile
  const int rowb1 = rowb0 + 16;                            // second 16-row tile
  const int c0    = n0 + wn * 32 + (lane & 15);
  const int c1    = c0 + 16;
  FAcc o00, o01, o10, o11;
  o00.v = acc00; o01.v = acc01; o10.v = acc10; o11.v = acc11;
#pragma unroll
  for (int rr = 0; rr < 8; ++rr) {
    const size_t r0 = (size_t)(rowb0 + rr);
    const size_t r1 = (size_t)(rowb1 + rr);
    if (c0 < N) {
      float v0 = o00.f[rr]; if (addsrc) v0 += addsrc[r0 * ldc + c0];
      C[r0 * ldc + c0] = v0;
      float v1 = o10.f[rr]; if (addsrc) v1 += addsrc[r1 * ldc + c0];
      C[r1 * ldc + c0] = v1;
    }
    if (c1 < N) {
      float v0 = o01.f[rr]; if (addsrc) v0 += addsrc[r0 * ldc + c1];
      C[r0 * ldc + c1] = v0;
      float v1 = o11.f[rr]; if (addsrc) v1 += addsrc[r1 * ldc + c1];
      C[r1 * ldc + c1] = v1;
    }
  }
}

// ---------------- causal depthwise conv (K=4) + SiLU on xBC slice of zxbcdt ----------------
__global__ void k_conv_silu(const float* __restrict__ zxbcdt, const float* __restrict__ cw,
                            const float* __restrict__ cb, float* __restrict__ xBCc) {
  int idx = blockIdx.x * blockDim.x + threadIdx.x;        // ROWS*CONVDIM
  if (idx >= ROWS * CONVDIM) return;
  int c = idx % CONVDIM;
  int l = (idx / CONVDIM) % SEQLEN;
  int b = idx / (CONVDIM * SEQLEN);
  float acc = cb[c];
#pragma unroll
  for (int k = 0; k < DCONV; ++k) {
    int ls = l + k - (DCONV - 1);
    if (ls >= 0)
      acc += zxbcdt[((size_t)(b * SEQLEN + ls)) * DINPROJ + DINNER + c] * cw[c * DCONV + k];
  }
  xBCc[idx] = siluf(acc);
}

// ---------------- dt = softplus(raw + bias), dA = exp(dt * -exp(A_log)) ----------------
__global__ void k_dt_prep(const float* __restrict__ zxbcdt, const float* __restrict__ dtb,
                          const float* __restrict__ A_log, float* __restrict__ dt,
                          float* __restrict__ dA) {
  int idx = blockIdx.x * blockDim.x + threadIdx.x;        // ROWS*NHEADS
  if (idx >= ROWS * NHEADS) return;
  int h = idx & (NHEADS - 1);
  int row = idx >> 3;
  float v = zxbcdt[(size_t)row * DINPROJ + (DINNER + CONVDIM) + h] + dtb[h];
  float sp = (v > 20.f) ? v : log1pf(expf(v));
  dt[idx] = sp;
  dA[idx] = expf(sp * (-expf(A_log[h])));
}

// ---------------- sequential SSD scan: one block per (b,h), LDS chunks of 32 steps ----------------
#define SCH 32
__global__ __launch_bounds__(256)
void k_ssd_scan(const float* __restrict__ xBCc, const float* __restrict__ dt,
                const float* __restrict__ dA, const float* __restrict__ Dv,
                float* __restrict__ y) {
  __shared__ float xs[SCH][64];
  __shared__ float Bsh[SCH][64];
  __shared__ float Csh[SCH][64];
  __shared__ float dts[SCH];
  __shared__ float dAs[SCH];

  const int b = blockIdx.x >> 3;
  const int h = blockIdx.x & 7;
  const int tid = threadIdx.x;
  const int p  = tid >> 2;            // headdim row 0..63
  const int ng = (tid & 3) * 16;      // 16-wide slice of state cols
  const float Dh = Dv[h];

  float hst[16];
#pragma unroll
  for (int i = 0; i < 16; ++i) hst[i] = 0.f;

  for (int c = 0; c < SEQLEN / SCH; ++c) {
    const int l0 = c * SCH;
    { // stage chunk: 2048 elems of each of x/B/C, 8 per thread
      const int e0 = tid * 8;
      const int t  = e0 >> 6, ch = e0 & 63;
      const float* rowp = &xBCc[((size_t)(b * SEQLEN + l0 + t)) * CONVDIM];
      *(float4*)&xs[t][ch]      = *(const float4*)&rowp[h * HEADDIM + ch];
      *(float4*)&xs[t][ch + 4]  = *(const float4*)&rowp[h * HEADDIM + ch + 4];
      *(float4*)&Bsh[t][ch]     = *(const float4*)&rowp[DINNER + ch];
      *(float4*)&Bsh[t][ch + 4] = *(const float4*)&rowp[DINNER + ch + 4];
      *(float4*)&Csh[t][ch]     = *(const float4*)&rowp[DINNER + DSTATE + ch];
      *(float4*)&Csh[t][ch + 4] = *(const float4*)&rowp[DINNER + DSTATE + ch + 4];
      if (tid < SCH) {
        dts[tid] = dt[((size_t)(b * SEQLEN + l0 + tid)) * NHEADS + h];
        dAs[tid] = dA[((size_t)(b * SEQLEN + l0 + tid)) * NHEADS + h];
      }
    }
    __syncthreads();

    for (int t = 0; t < SCH; ++t) {
      const float dtv = dts[t], dav = dAs[t];
      const float xv = xs[t][p];
      const float coef = dtv * xv;
      float part = 0.f;
#pragma unroll
      for (int i = 0; i < 16; ++i) {
        hst[i] = hst[i] * dav + coef * Bsh[t][ng + i];
        part += hst[i] * Csh[t][ng + i];
      }
      part += __shfl_xor(part, 1);
      part += __shfl_xor(part, 2);
      if ((tid & 3) == 0)
        y[((size_t)(b * SEQLEN + l0 + t)) * DINNER + h * HEADDIM + p] = part + Dh * xv;
    }
    __syncthreads();
  }
}

// ---------------- gated RMSNorm -> bf16 ----------------
__global__ __launch_bounds__(256)
void k_gate_norm(const float* __restrict__ y, const float* __restrict__ zxbcdt,
                 const float* __restrict__ nw, unsigned short* __restrict__ yn) {
  __shared__ float gbuf[DINNER];
  __shared__ float red[8];
  const int row = blockIdx.x;
  const int tid = threadIdx.x;
  float acc = 0.f;
#pragma unroll
  for (int j = 0; j < 2; ++j) {
    int e = tid + j * 256;
    float yv = y[(size_t)row * DINNER + e];
    float zv = zxbcdt[(size_t)row * DINPROJ + e];
    float g = yv * siluf(zv);
    gbuf[e] = g;
    acc += g * g;
  }
#pragma unroll
  for (int o = 16; o > 0; o >>= 1) acc += __shfl_xor(acc, o);
  if ((tid & 31) == 0) red[tid >> 5] = acc;
  __syncthreads();
  if (tid == 0) {
    float s = 0.f;
#pragma unroll
    for (int w = 0; w < 8; ++w) s += red[w];
    red[0] = rsqrtf(s / (float)DINNER + EPSV);
  }
  __syncthreads();
  const float rstd = red[0];
#pragma unroll
  for (int j = 0; j < 2; ++j) {
    int e = tid + j * 256;
    yn[(size_t)row * DINNER + e] = f2bf_bits(gbuf[e] * rstd * nw[e]);
  }
}

// ---------------- pooling: (mean + max) * 0.5 ----------------
__global__ void k_pool(const float* __restrict__ x, float* __restrict__ pooled) {
  const int b = blockIdx.x, d = threadIdx.x;
  float mx = -3.4e38f, sm = 0.f;
  for (int l = 0; l < SEQLEN; ++l) {
    float v = x[((size_t)(b * SEQLEN + l)) * DMODEL + d];
    sm += v; mx = fmaxf(mx, v);
  }
  pooled[b * DMODEL + d] = 0.5f * (sm / (float)SEQLEN + mx);
}

// ---------------- classification head (single block) ----------------
__global__ __launch_bounds__(256)
void k_head(const float* __restrict__ pooled,
            const float* __restrict__ pw, const float* __restrict__ pb,
            const float* __restrict__ c1w, const float* __restrict__ c1b,
            const float* __restrict__ c2w, const float* __restrict__ c2b,
            float* __restrict__ out) {
  __shared__ float ps[BATCH * DMODEL];
  __shared__ float p1[BATCH * DMODEL];
  __shared__ float h1[BATCH * 128];
  const int tid = threadIdx.x;
  for (int i = tid; i < BATCH * DMODEL; i += 256) ps[i] = pooled[i];
  __syncthreads();
  for (int i = tid; i < BATCH * DMODEL; i += 256) {
    int b = i >> 8, o = i & 255;
    float acc = pb[o];
    for (int k = 0; k < DMODEL; ++k) acc += ps[b * DMODEL + k] * pw[o * DMODEL + k];
    p1[i] = geluf(acc);
  }
  __syncthreads();
  for (int i = tid; i < BATCH * 128; i += 256) {
    int b = i >> 7, o = i & 127;
    float acc = c1b[o];
    for (int k = 0; k < DMODEL; ++k) acc += p1[b * DMODEL + k] * c1w[o * DMODEL + k];
    h1[i] = geluf(acc);
  }
  __syncthreads();
  if (tid < BATCH * 2) {
    int b = tid >> 1, o = tid & 1;
    float acc = c2b[o];
    for (int k = 0; k < 128; ++k) acc += h1[b * 128 + k] * c2w[o * 128 + k];
    out[b * 2 + o] = acc;
  }
}

extern "C" void kernel_launch(void* const* d_in, const int* in_sizes, int n_in,
                              void* d_out, int out_size, void* d_ws, size_t ws_size,
                              hipStream_t stream) {
  const int*   ids   = (const int*)  d_in[0];
  const float* emb   = (const float*)d_in[1];
  const float* inw   = (const float*)d_in[2];
  const float* cw    = (const float*)d_in[3];
  const float* cb    = (const float*)d_in[4];
  const float* dtb   = (const float*)d_in[5];
  const float* alog  = (const float*)d_in[6];
  const float* Dv    = (const float*)d_in[7];
  const float* nw    = (const float*)d_in[8];
  const float* ow    = (const float*)d_in[9];
  const float* pw    = (const float*)d_in[10];
  const float* pb    = (const float*)d_in[11];
  const float* c1w   = (const float*)d_in[12];
  const float* c1b   = (const float*)d_in[13];
  const float* c2w   = (const float*)d_in[14];
  const float* c2b   = (const float*)d_in[15];
  float* out = (float*)d_out;

  char* w = (char*)d_ws;
  float*          x      = (float*)(w);                          // 8 MB
  unsigned short* xb     = (unsigned short*)(w + 8388608);       // 4 MB
  float*          zxbcdt = (float*)(w + 12582912);               // 38 MB
  float*          xBCc   = (float*)(w + 50593792);               // 21 MB
  float*          dtbuf  = (float*)(w + 71565312);               // 256 KB
  float*          dAbuf  = (float*)(w + 71827456);               // 256 KB
  float*          ybuf   = (float*)(w + 72089600);               // 16 MB
  unsigned short* ynb    = (unsigned short*)(w + 88866816);      // 8 MB
  unsigned short* wA     = (unsigned short*)(w + 97255424);      // 580 KB
  unsigned short* wO     = (unsigned short*)(w + 97849344);      // 256 KB
  float*          pooled = (float*)(w + 98111488);               // 8 KB

  // embedding + bf16 shadow
  k_embed<<<(ROWS * DMODEL) / 256, 256, 0, stream>>>(ids, emb, x, xb);

  for (int layer = 0; layer < NLAYERS; ++layer) {
    const float* inw_l  = inw  + (size_t)layer * DINPROJ * DMODEL;
    const float* cw_l   = cw   + (size_t)layer * CONVDIM * DCONV;
    const float* cb_l   = cb   + (size_t)layer * CONVDIM;
    const float* dtb_l  = dtb  + (size_t)layer * NHEADS;
    const float* alog_l = alog + (size_t)layer * NHEADS;
    const float* Dv_l   = Dv   + (size_t)layer * NHEADS;
    const float* nw_l   = nw   + (size_t)layer * DINNER;
    const float* ow_l   = ow   + (size_t)layer * DMODEL * DINNER;

    // weights -> bf16
    k_f32_to_bf16<<<(DINPROJ * DMODEL + 255) / 256, 256, 0, stream>>>(inw_l, wA, DINPROJ * DMODEL);
    k_f32_to_bf16<<<(DMODEL * DINNER + 255) / 256, 256, 0, stream>>>(ow_l, wO, DMODEL * DINNER);

    // in_proj: zxbcdt[ROWS,1160] = xb[ROWS,256] @ wA[1160,256]^T
    {
      dim3 grid((DINPROJ + 63) / 64, ROWS / 128);
      k_gemm_bf16<<<grid, 256, 0, stream>>>(xb, wA, nullptr, zxbcdt,
                                            ROWS, DINPROJ, DMODEL, DINPROJ);
    }

    k_conv_silu<<<(ROWS * CONVDIM + 255) / 256, 256, 0, stream>>>(zxbcdt, cw_l, cb_l, xBCc);
    k_dt_prep<<<(ROWS * NHEADS + 255) / 256, 256, 0, stream>>>(zxbcdt, dtb_l, alog_l, dtbuf, dAbuf);
    k_ssd_scan<<<BATCH * NHEADS, 256, 0, stream>>>(xBCc, dtbuf, dAbuf, Dv_l, ybuf);
    k_gate_norm<<<ROWS, 256, 0, stream>>>(ybuf, zxbcdt, nw_l, ynb);

    // out_proj + residual: x[ROWS,256] += ynb[ROWS,512] @ wO[256,512]^T
    {
      dim3 grid(DMODEL / 64, ROWS / 128);
      k_gemm_bf16<<<grid, 256, 0, stream>>>(ynb, wO, x, x,
                                            ROWS, DMODEL, DINNER, DMODEL);
    }
    if (layer + 1 < NLAYERS)
      k_f32_to_bf16<<<(ROWS * DMODEL + 255) / 256, 256, 0, stream>>>(x, xb, ROWS * DMODEL);
  }

  k_pool<<<BATCH, DMODEL, 0, stream>>>(x, pooled);
  k_head<<<1, 256, 0, stream>>>(pooled, pw, pb, c1w, c1b, c2w, c2b, out);
}
